// lstm_gcn_81724637708457
// MI455X (gfx1250) — matrix operand order repaired and checked by reading, not compiled, not run on hardware
//
#include <hip/hip_runtime.h>
#include <cstdint>
#include <cstddef>

typedef __attribute__((ext_vector_type(16))) __bf16 v16bf;
typedef __attribute__((ext_vector_type(8)))  float  v8f;

#define SEQ  16
#define NN   50000
#define DD   128
#define EE   1600000
#define G4   512     // 4*D (gate width)
#define ROWT 16      // rows of the node dimension per block
#define KTN  4       // number of K=32 tiles along D=128

__device__ __forceinline__ float sigmoidf_(float x) { return 1.0f / (1.0f + expf(-x)); }

// Async-copy one contiguous 8KB x-tile (16 rows x 128 f32) into LDS.
// 128 threads x 4 iterations x 16B = 8192B; each wave issues 4 async instrs.
__device__ __forceinline__ void async_tile_load(const float* __restrict__ gbase,
                                                const float* ldsDst, int tid) {
#pragma unroll
  for (int i = 0; i < 4; ++i) {
    int f = tid + i * 128;                                 // float4 index
    unsigned goff = (unsigned)(f * 16);                    // byte offset in tile
    unsigned lds  = (unsigned)(uintptr_t)(ldsDst + f * 4); // LDS byte offset = addr[31:0]
    asm volatile("global_load_async_to_lds_b128 %0, %1, %2"
                 :: "v"(lds), "v"(goff), "s"(gbase) : "memory");
  }
}

// ---------------------------------------------------------------------------
// Pack a weight matrix into bf16 WMMA B-fragment order:
//   dst[((ntile*KT + kt)*32 + lane)*16 + e], col = ntile*16 + lane%16,
//   k = kt*32 + (lane/16)*8 + (e&7) + (e/8)*16   (B 32x16: lanes 0-15 K 0-7,16-23; lanes 16-31 K 8-15,24-31)
// transposed=1: src is [Ncols][K] (PyTorch W_ih/W_hh, used as x @ W^T)
// transposed=0: src is [K][Ncols] (GCN W, used as x @ W)
// ---------------------------------------------------------------------------
__global__ void k_pack_b(const float* __restrict__ src, __bf16* __restrict__ dst,
                         int Ncols, int K, int transposed) {
  int idx = blockIdx.x * blockDim.x + threadIdx.x;
  if (idx >= Ncols * K) return;
  int e    = idx & 15;
  int lane = (idx >> 4) & 31;
  int tile = idx >> 9;
  int kts  = K >> 5;
  int kt   = tile % kts;
  int nt   = tile / kts;
  int n = nt * 16 + (lane & 15);
  int k = kt * 32 + ((lane >> 4) << 3) + (e & 7) + ((e & 8) << 1);
  float v = transposed ? src[n * K + k] : src[k * Ncols + n];
  dst[idx] = (__bf16)v;
}

__global__ void k_bias(const float* __restrict__ a, const float* __restrict__ b,
                       float* __restrict__ o) {
  int i = blockIdx.x * blockDim.x + threadIdx.x;
  if (i < G4) o[i] = a[i] + b[i];
}

// ---------------------------------------------------------------------------
// Persistent LSTM: block = 128 threads (4 waves) owns 16 nodes for all 16 steps.
// x tiles stream in via double-buffered GLOBAL_LOAD_ASYNC_TO_LDS (ASYNCcnt),
// hidden-state recurrence stays in LDS in bf16 A-fragment layout, gates via
// v_wmma_f32_16x16x32_bf16 with two interleaved accumulator chains per wave.
// Bias is hoisted into per-lane registers before the time loop.
// ---------------------------------------------------------------------------
__global__ __launch_bounds__(128) void k_lstm(const float* __restrict__ x,
                                              const __bf16* __restrict__ wihP,
                                              const __bf16* __restrict__ whhP,
                                              const float* __restrict__ bias,
                                              float* __restrict__ hout) {
  __shared__ __attribute__((aligned(16))) float  xRaw[2][ROWT * DD]; // 2 x 8KB
  __shared__ __attribute__((aligned(32))) __bf16 xA[KTN * 32 * 16];
  __shared__ __attribute__((aligned(32))) __bf16 hA[KTN * 32 * 16];
  __shared__ float gates[ROWT * G4];
  __shared__ float cSt[ROWT * DD];

  const int tid  = threadIdx.x;
  const int lane = tid & 31;
  const int wv   = tid >> 5;
  const int r0   = blockIdx.x * ROWT;

  // prologue: kick off async copy of x_0
  async_tile_load(x + (size_t)r0 * DD, &xRaw[0][0], tid);

  // hoist loop-invariant per-lane bias values into registers
  float bvA[4], bvB[4];
#pragma unroll
  for (int ctp = 0; ctp < 4; ++ctp) {
    bvA[ctp] = bias[(wv + ctp * 8) * 16 + (lane & 15)];
    bvB[ctp] = bias[(wv + ctp * 8 + 4) * 16 + (lane & 15)];
  }

  for (int i = tid; i < KTN * 32 * 16; i += 128) hA[i] = (__bf16)0.0f;
  for (int i = tid; i < ROWT * DD;    i += 128) cSt[i] = 0.0f;

  for (int t = 0; t < SEQ; ++t) {
    // prefetch next tile, then drain the current one (loads retire in order)
    if (t + 1 < SEQ) {
      async_tile_load(x + ((size_t)(t + 1) * NN + r0) * DD, &xRaw[(t + 1) & 1][0], tid);
      asm volatile("s_wait_asynccnt 0x4" ::: "memory");
    } else {
      asm volatile("s_wait_asynccnt 0x0" ::: "memory");
    }
    __syncthreads();

    // convert current raw tile into A-fragment layout (f32 -> bf16)
    const float* xr = &xRaw[t & 1][0];
    for (int idx = tid; idx < KTN * 32 * 16; idx += 128) {
      int e = idx & 15, ln = (idx >> 4) & 31, kt = idx >> 9;
      int row = ln & 15;
      int k = kt * 32 + ((ln & 16) >> 1) + (e & 7) + ((e & 8) << 1);
      xA[idx] = (__bf16)xr[row * DD + k];
    }
    __syncthreads();

    v16bf ax[KTN], ah[KTN];
#pragma unroll
    for (int kt = 0; kt < KTN; ++kt) {
      ax[kt] = *(const v16bf*)&xA[(kt * 32 + lane) * 16];
      ah[kt] = *(const v16bf*)&hA[(kt * 32 + lane) * 16];
    }

    // 8 gate column-tiles per wave, processed as 4 pairs of independent chains
#pragma unroll
    for (int ctp = 0; ctp < 4; ++ctp) {
      const int ctA = wv + ctp * 8;
      const int ctB = ctA + 4;
      v8f acc0, acc1;
#pragma unroll
      for (int r = 0; r < 8; ++r) { acc0[r] = bvA[ctp]; acc1[r] = bvB[ctp]; }
#pragma unroll
      for (int kt = 0; kt < KTN; ++kt) {
        v16bf b0 = *(const v16bf*)&wihP[(((size_t)ctA * KTN + kt) * 32 + lane) * 16];
        v16bf b1 = *(const v16bf*)&wihP[(((size_t)ctB * KTN + kt) * 32 + lane) * 16];
        acc0 = __builtin_amdgcn_wmma_f32_16x16x32_bf16(false, ax[kt], false, b0,
                                                       (short)0, acc0, false, false);
        acc1 = __builtin_amdgcn_wmma_f32_16x16x32_bf16(false, ax[kt], false, b1,
                                                       (short)0, acc1, false, false);
      }
#pragma unroll
      for (int kt = 0; kt < KTN; ++kt) {
        v16bf b0 = *(const v16bf*)&whhP[(((size_t)ctA * KTN + kt) * 32 + lane) * 16];
        v16bf b1 = *(const v16bf*)&whhP[(((size_t)ctB * KTN + kt) * 32 + lane) * 16];
        acc0 = __builtin_amdgcn_wmma_f32_16x16x32_bf16(false, ah[kt], false, b0,
                                                       (short)0, acc0, false, false);
        acc1 = __builtin_amdgcn_wmma_f32_16x16x32_bf16(false, ah[kt], false, b1,
                                                       (short)0, acc1, false, false);
      }
      const int col   = lane & 15;
      const int rbase = (lane & 16) >> 1;   // C/D: lanes 16-31 hold M = r+8
#pragma unroll
      for (int r = 0; r < 8; ++r) {
        gates[(r + rbase) * G4 + ctA * 16 + col] = acc0[r];
        gates[(r + rbase) * G4 + ctB * 16 + col] = acc1[r];
      }
    }
    __syncthreads();

    // elementwise cell update; write h back into A-frag layout for next step
    for (int j = tid; j < ROWT * DD; j += 128) {
      int row = j >> 7, hc = j & 127;
      float gi = sigmoidf_(gates[row * G4 + hc]);
      float gf = sigmoidf_(gates[row * G4 + 128 + hc]);
      float gg = tanhf(gates[row * G4 + 256 + hc]);
      float go = sigmoidf_(gates[row * G4 + 384 + hc]);
      float c  = gf * cSt[j] + gi * gg;
      cSt[j]   = c;
      float h  = go * tanhf(c);
      int kt = hc >> 5, kk = hc & 31;
      int ln = (((kk >> 3) & 1) << 4) + row;
      int e  = (kk & 7) | ((kk & 16) >> 1);
      hA[(kt * 32 + ln) * 16 + e] = (__bf16)h;
      if (t == SEQ - 1) hout[(size_t)(r0 + row) * DD + hc] = h;
    }
    __syncthreads();
  }
}

// ---------------------------------------------------------------------------
// y[N x 128] = x[N x 128] @ Wp (bf16 WMMA, f32 accumulate). 4 waves, 16 rows.
// ---------------------------------------------------------------------------
__global__ __launch_bounds__(128) void k_gemm(const float* __restrict__ x,
                                              const __bf16* __restrict__ wP,
                                              float* __restrict__ y) {
  __shared__ __attribute__((aligned(32))) __bf16 xA[KTN * 32 * 16];
  const int tid = threadIdx.x, lane = tid & 31, wv = tid >> 5;
  const int r0 = blockIdx.x * ROWT;

  for (int idx = tid; idx < KTN * 32 * 16; idx += 128) {
    int e = idx & 15, ln = (idx >> 4) & 31, kt = idx >> 9;
    int row = ln & 15;
    int k = kt * 32 + ((ln & 16) >> 1) + (e & 7) + ((e & 8) << 1);
    xA[idx] = (__bf16)x[(size_t)(r0 + row) * DD + k];
  }
  __syncthreads();

  v16bf ax[KTN];
#pragma unroll
  for (int kt = 0; kt < KTN; ++kt) ax[kt] = *(const v16bf*)&xA[(kt * 32 + lane) * 16];

  const int ctA = wv, ctB = wv + 4;
  v8f acc0 = {}, acc1 = {};
#pragma unroll
  for (int kt = 0; kt < KTN; ++kt) {
    v16bf b0 = *(const v16bf*)&wP[(((size_t)ctA * KTN + kt) * 32 + lane) * 16];
    v16bf b1 = *(const v16bf*)&wP[(((size_t)ctB * KTN + kt) * 32 + lane) * 16];
    acc0 = __builtin_amdgcn_wmma_f32_16x16x32_bf16(false, ax[kt], false, b0,
                                                   (short)0, acc0, false, false);
    acc1 = __builtin_amdgcn_wmma_f32_16x16x32_bf16(false, ax[kt], false, b1,
                                                   (short)0, acc1, false, false);
  }
  const int col = lane & 15;
  const int rbase = (lane & 16) >> 1;
#pragma unroll
  for (int r = 0; r < 8; ++r) {
    y[(size_t)(r0 + r + rbase) * DD + ctA * 16 + col] = acc0[r];
    y[(size_t)(r0 + r + rbase) * DD + ctB * 16 + col] = acc1[r];
  }
}

__global__ void k_init(float* __restrict__ o, const float* __restrict__ b) {
  size_t i = (size_t)blockIdx.x * blockDim.x + threadIdx.x;
  if (i < (size_t)NN * DD) o[i] = b[i & 127];
}

__global__ void k_relu(float* __restrict__ o) {
  size_t i = (size_t)blockIdx.x * blockDim.x + threadIdx.x;
  if (i < (size_t)NN * DD) o[i] = fmaxf(o[i], 0.0f);
}

// one thread = one edge * 4 features (float4 gather, 4 atomic f32 adds)
__global__ void k_scatter(const float* __restrict__ xw,
                          const long long* __restrict__ edges,
                          float* __restrict__ out) {
  size_t i = (size_t)blockIdx.x * blockDim.x + threadIdx.x;
  if (i >= (size_t)EE * 32) return;
  size_t e = i >> 5;
  int    c = (int)(i & 31) * 4;
  long long s = edges[e];
  long long d = edges[(size_t)EE + e];
  const float4 v = *(const float4*)&xw[(size_t)s * DD + c];
  float* op = &out[(size_t)d * DD + c];
  atomicAdd(op + 0, v.x);
  atomicAdd(op + 1, v.y);
  atomicAdd(op + 2, v.z);
  atomicAdd(op + 3, v.w);
}

extern "C" void kernel_launch(void* const* d_in, const int* in_sizes, int n_in,
                              void* d_out, int out_size, void* d_ws, size_t ws_size,
                              hipStream_t stream) {
  // setup_inputs() dict order
  const float*     x     = (const float*)d_in[0];      // [16, 50000, 128]
  const long long* edges = (const long long*)d_in[1];  // [2, E] int64
  const float*     Wih   = (const float*)d_in[2];      // [512, 128]
  const float*     Whh   = (const float*)d_in[3];      // [512, 128]
  const float*     bih   = (const float*)d_in[4];      // [512]
  const float*     bhh   = (const float*)d_in[5];      // [512]
  const float*     W1    = (const float*)d_in[6];      // [128, 128]
  const float*     b1    = (const float*)d_in[7];      // [128]
  const float*     W2    = (const float*)d_in[8];      // [128, 128]
  const float*     b2    = (const float*)d_in[9];      // [128]
  float* out = (float*)d_out;                          // [50000, 128]

  char* ws = (char*)d_ws;
  size_t off = 0;
  __bf16* wihP = (__bf16*)(ws + off); off += (size_t)G4 * DD * 2;   // 128 KB
  __bf16* whhP = (__bf16*)(ws + off); off += (size_t)G4 * DD * 2;   // 128 KB
  __bf16* w1P  = (__bf16*)(ws + off); off += (size_t)DD * DD * 2;   //  32 KB
  __bf16* w2P  = (__bf16*)(ws + off); off += (size_t)DD * DD * 2;   //  32 KB
  float*  bsum = (float*)(ws + off);  off += (size_t)G4 * 4;
  off = (off + 255) & ~(size_t)255;
  float* bufA = (float*)(ws + off); off += (size_t)NN * DD * 4;     // 25.6 MB
  float* bufB = (float*)(ws + off); off += (size_t)NN * DD * 4;     // 25.6 MB

  const int NB  = NN / ROWT;                              // 3125 row-tile blocks
  const int NEL = (NN * DD + 255) / 256;                  // elementwise grid
  const int NSC = (int)(((size_t)EE * 32 + 255) / 256);   // 200000 scatter blocks

  // 1) pack weights / fuse bias
  k_pack_b<<<(G4 * DD + 255) / 256, 256, 0, stream>>>(Wih, wihP, G4, DD, 1);
  k_pack_b<<<(G4 * DD + 255) / 256, 256, 0, stream>>>(Whh, whhP, G4, DD, 1);
  k_pack_b<<<(DD * DD + 255) / 256, 256, 0, stream>>>(W1, w1P, DD, DD, 0);
  k_pack_b<<<(DD * DD + 255) / 256, 256, 0, stream>>>(W2, w2P, DD, DD, 0);
  k_bias<<<(G4 + 255) / 256, 256, 0, stream>>>(bih, bhh, bsum);

  // 2) persistent WMMA LSTM (async x streaming) -> h in bufA
  k_lstm<<<NB, 128, 0, stream>>>(x, wihP, whhP, bsum, bufA);

  // 3) GCN layer 1: xw = h @ W1 (bufB); out1 = b1 + scatter(xw[src] -> dst); relu
  k_gemm<<<NB, 128, 0, stream>>>(bufA, w1P, bufB);
  k_init<<<NEL, 256, 0, stream>>>(bufA, b1);           // reuse bufA as out1
  k_scatter<<<NSC, 256, 0, stream>>>(bufB, edges, bufA);
  k_relu<<<NEL, 256, 0, stream>>>(bufA);

  // 4) GCN layer 2 -> d_out
  k_gemm<<<NB, 128, 0, stream>>>(bufA, w2P, bufB);
  k_init<<<NEL, 256, 0, stream>>>(out, b2);
  k_scatter<<<NSC, 256, 0, stream>>>(bufB, edges, out);
}
